// PretrainDecoder_21294447853704
// MI455X (gfx1250) — matrix operand order repaired and checked by reading, not compiled, not run on hardware
//
#include <hip/hip_runtime.h>
#include <hip/hip_bf16.h>

// ---------------------------------------------------------------------------
// Residual VQ (4 stages, B=65536, D=512, K=1024) + 3-layer MLP decoder.
// bf16 WMMA (16x16x32) everywhere, f32 accumulate.
// VQ codebook tiles now streamed with CDNA5 async loads
// (global_load_async_to_lds_b128 + ASYNCcnt) and double-buffered in LDS.
// ---------------------------------------------------------------------------

typedef __attribute__((ext_vector_type(16))) __bf16 v16bf;
typedef __attribute__((ext_vector_type(8)))  float  v8f;

union FragU { v16bf v; __bf16 h[16]; uint4 q[2]; };   // 32 bytes = 8 VGPRs
union HalfU { __bf16 h[8]; uint4 q; };                // 16 bytes

#define B_ROWS 65536
#define DIM    512
#define GQ     4
#define KC     1024
#define HID    128
#define AOUT   7

__device__ __forceinline__ v8f wmma_bf16(const FragU& a, const FragU& b, v8f c) {
  return __builtin_amdgcn_wmma_f32_16x16x32_bf16(
      /*neg_a=*/false, a.v, /*neg_b=*/false, b.v,
      /*c_mod=*/(short)0, c, /*reuse_a=*/false, /*reuse_b=*/false);
}

// Async memory->LDS 16B copy (per lane), tracked with ASYNCcnt.
__device__ __forceinline__ void async_copy16(unsigned int lds_byte_addr,
                                             unsigned long long gaddr) {
  asm volatile("global_load_async_to_lds_b128 %0, %1, off"
               :: "v"(lds_byte_addr), "v"(gaddr) : "memory");
}
__device__ __forceinline__ void wait_async0() {
  asm volatile("s_wait_asynccnt 0" ::: "memory");
}

// ---------------------------- prep kernels ---------------------------------

__global__ void k_cvt_cb(const float* __restrict__ src, __bf16* __restrict__ dst) {
  size_t i = (size_t)blockIdx.x * blockDim.x + threadIdx.x;
  dst[i] = (__bf16)src[i];
}

__global__ void k_cnorm(const float* __restrict__ cb, float* __restrict__ cn) {
  int c = blockIdx.x * blockDim.x + threadIdx.x;   // 0..GQ*KC-1
  const float* row = cb + (size_t)c * DIM;
  float s = 0.f;
  for (int d = 0; d < DIM; ++d) { float v = row[d]; s += v * v; }
  cn[c] = 0.5f * s;
}

__global__ void k_cvt_w(const float* __restrict__ w1, const float* __restrict__ w2,
                        const float* __restrict__ w3,
                        __bf16* __restrict__ w1t, __bf16* __restrict__ w2t,
                        __bf16* __restrict__ w3t) {
  int i = blockIdx.x * blockDim.x + threadIdx.x;
  if (i < HID * DIM) {                      // w1t[h][d] = w1[d][h]
    int hh = i / DIM, d = i % DIM;
    w1t[i] = (__bf16)w1[d * HID + hh];
  } else if (i < HID * DIM + HID * HID) {   // w2t[n][k] = w2[k][n]
    int j = i - HID * DIM;
    int n = j / HID, k = j % HID;
    w2t[j] = (__bf16)w2[k * HID + n];
  } else {                                  // w3t[n][k] = w3[k][n], pad n>=7 with 0
    int j = i - HID * DIM - HID * HID;      // 0..16*128-1
    int n = j / HID, k = j % HID;
    w3t[j] = (n < AOUT) ? (__bf16)w3[k * AOUT + n] : (__bf16)0.f;
  }
}

// ------------------------------ VQ kernel ----------------------------------
// grid 512 x 256 threads (8 waves). Wave w owns rows [blk*128 + w*16, +16).
// A-fragment layout (16-bit A 16x32): lane L -> row = L&15;
//   elems 0..7  = K 8*(L>>4)+0..7 ; elems 8..15 = K 16+8*(L>>4)+0..7.
// B-fragment layout (16-bit B 32x16): lane L -> col = L&15;
//   elems 0..7 = K 8*(L>>4)+0..7 ; elems 8..15 = K 16+8*(L>>4)+0..7 (two 16B pieces of a codebook row).
// 16-code tiles double-buffered in LDS via async loads.

#define LSTRIDE 520   // 512 + 8 pad (elements): 1040B rows, 16B aligned

__global__ __launch_bounds__(256)
void vq_kernel(const float* __restrict__ z, const __bf16* __restrict__ cb,
               const float* __restrict__ cnorm, __bf16* __restrict__ st) {
  __shared__ __align__(16) __bf16 btile[2][16 * LSTRIDE];  // 2 x 16.25 KB
  __shared__ int idxsh[8][16];

  const int tid  = threadIdx.x;
  const int lane = tid & 31;
  const int w    = tid >> 5;
  const int hh   = lane >> 4;     // lane half
  const int l16  = lane & 15;
  const int kb   = hh * 16;
  const size_t row = (size_t)blockIdx.x * 128 + w * 16 + l16;
  const float* zrow = z + row * DIM;

  const unsigned int lds0 = (unsigned int)(uintptr_t)&btile[0][0];
  const unsigned int lds1 = (unsigned int)(uintptr_t)&btile[1][0];
  // Per-thread chunk assignment for tile staging: 1024 16B chunks / 256 thr.
  // chunk ci = j*256 + tid  ->  row r = ci>>6, 16B-chunk c = ci&63.
  const int r0 = tid >> 6, c0 = tid & 63;

  // Residual A-fragments in registers (initialized from z, f32 -> bf16)
  FragU rA[16];
#pragma unroll
  for (int f = 0; f < 16; ++f) {
    const int d0 = f * 32 + hh * 8;
#pragma unroll
    for (int e = 0; e < 8; ++e) {
      rA[f].h[e]     = (__bf16)zrow[d0 + e];
      rA[f].h[e + 8] = (__bf16)zrow[d0 + 16 + e];
    }
  }

  for (int g = 0; g < GQ; ++g) {
    const __bf16* cbg = cb + (size_t)g * KC * DIM;
    float best[8]; int bidx[8];
#pragma unroll
    for (int i = 0; i < 8; ++i) { best[i] = -3.4e38f; bidx[i] = 0; }

    // Prologue: async-load tile 0 into buffer 0.
#pragma unroll
    for (int j = 0; j < 4; ++j) {
      const int r = r0 + j * 4;                       // 4 rows apart per j
      const unsigned long long ga =
          (unsigned long long)(uintptr_t)(cbg + (size_t)r * DIM + c0 * 8);
      async_copy16(lds0 + (unsigned int)(r * LSTRIDE + c0 * 8) * 2u, ga);
    }

    for (int nt = 0; nt < KC / 16; ++nt) {            // 64 16-code tiles
      const int cur = nt & 1;
      wait_async0();            // my async writes for buf[cur] are done
      __syncthreads();          // everyone's writes are done

      // Prefetch next tile into the other buffer while we compute.
      if (nt + 1 < KC / 16) {
        const unsigned int ldsn = (cur ? lds0 : lds1);
#pragma unroll
        for (int j = 0; j < 4; ++j) {
          const int r = r0 + j * 4;
          const unsigned long long ga = (unsigned long long)(uintptr_t)(
              cbg + (size_t)((nt + 1) * 16 + r) * DIM + c0 * 8);
          async_copy16(ldsn + (unsigned int)(r * LSTRIDE + c0 * 8) * 2u, ga);
        }
      }

      const __bf16* bbase = &btile[cur][0];
      v8f accL = {}; v8f accH = {};
#pragma unroll
      for (int ds = 0; ds < 8; ++ds) {                // two independent chains
        FragU bL, bH;
        const uint4* pL = (const uint4*)(bbase + l16 * LSTRIDE + ds * 32 + hh * 8);
        const uint4* pH = (const uint4*)(bbase + l16 * LSTRIDE + (ds + 8) * 32 + hh * 8);
        bL.q[0] = pL[0]; bL.q[1] = pL[2];
        bH.q[0] = pH[0]; bH.q[1] = pH[2];
        accL = wmma_bf16(rA[ds],     bL, accL);
        accH = wmma_bf16(rA[ds + 8], bH, accH);
      }

      const float cn = cnorm[g * KC + nt * 16 + l16];
      const int n = nt * 16 + l16;
#pragma unroll
      for (int i = 0; i < 8; ++i) {
        const float s = accL[i] + accH[i] - cn;
        if (s > best[i]) { best[i] = s; bidx[i] = n; }
      }
    }

    // Argmax reduction across the 16 lanes of each half (C layout: lanes 0-15
    // hold rows 0-7, lanes 16-31 hold rows 8-15).
#pragma unroll
    for (int off = 8; off >= 1; off >>= 1) {
#pragma unroll
      for (int i = 0; i < 8; ++i) {
        const float ov = __shfl_xor(best[i], off, 16);
        const int   oi = __shfl_xor(bidx[i], off, 16);
        if (ov > best[i]) { best[i] = ov; bidx[i] = oi; }
      }
    }
    if (l16 == 0) {
#pragma unroll
      for (int i = 0; i < 8; ++i) idxsh[w][hh * 8 + i] = bidx[i];
    }
    __syncthreads();   // also guarantees all tile reads finished before reuse

    // Gather winning codeword and update residual fragments in place.
    const int code = idxsh[w][l16];
    const __bf16* crow = cbg + (size_t)code * DIM;
#pragma unroll
    for (int f = 0; f < 16; ++f) {
      const int d0 = f * 32 + hh * 8;
      HalfU q0, q1;
      q0.q = *(const uint4*)(crow + d0);
      q1.q = *(const uint4*)(crow + d0 + 16);
#pragma unroll
      for (int e = 0; e < 8; ++e) {
        rA[f].h[e]     = (__bf16)((float)rA[f].h[e]     - (float)q0.h[e]);
        rA[f].h[e + 8] = (__bf16)((float)rA[f].h[e + 8] - (float)q1.h[e]);
      }
    }
  }

  // st = z_vq = z - residual_final, stored bf16 in WMMA-friendly row-major.
  __bf16* strow = st + row * DIM;
#pragma unroll
  for (int f = 0; f < 16; ++f) {
    const int d0 = f * 32 + hh * 8;
    HalfU s0, s1;
#pragma unroll
    for (int e = 0; e < 8; ++e) {
      s0.h[e] = (__bf16)(zrow[d0 + e]      - (float)rA[f].h[e]);
      s1.h[e] = (__bf16)(zrow[d0 + 16 + e] - (float)rA[f].h[e + 8]);
    }
    *(uint4*)(strow + d0)      = s0.q;
    *(uint4*)(strow + d0 + 16) = s1.q;
  }
}

// ------------------------------ MLP kernel ---------------------------------

#define HS 136   // LDS row stride (elements) for hidden activations

__global__ __launch_bounds__(256)
void mlp_kernel(const __bf16* __restrict__ st,
                const __bf16* __restrict__ w1t, const float* __restrict__ b1,
                const __bf16* __restrict__ w2t, const float* __restrict__ b2,
                const __bf16* __restrict__ w3t, const float* __restrict__ b3,
                float* __restrict__ out) {
  __shared__ __align__(16) __bf16 hbuf[8 * 16 * HS];   // ~35 KB, per-wave slices

  const int tid  = threadIdx.x;
  const int lane = tid & 31;
  const int w    = tid >> 5;
  const int hh   = lane >> 4;
  const int l16  = lane & 15;
  const int kb   = hh * 16;
  const size_t rbase = (size_t)blockIdx.x * 128 + w * 16;
  const size_t row   = rbase + l16;
  __bf16* hw = hbuf + (w * 16) * HS;

  // ---- layer 1: h1 = relu(st @ w1 + b1), K=512 ----
  FragU a1[16];
  const __bf16* srow = st + row * DIM;
#pragma unroll
  for (int f = 0; f < 16; ++f) {
    const int d0 = f * 32 + hh * 8;
    a1[f].q[0] = *(const uint4*)(srow + d0);
    a1[f].q[1] = *(const uint4*)(srow + d0 + 16);
  }
#pragma unroll
  for (int nt = 0; nt < 8; ++nt) {
    v8f acc = {};
#pragma unroll
    for (int ds = 0; ds < 16; ++ds) {
      FragU b;
      const __bf16* p = w1t + (size_t)(nt * 16 + l16) * DIM + ds * 32 + hh * 8;
      b.q[0] = *(const uint4*)p; b.q[1] = *(const uint4*)(p + 16);
      acc = wmma_bf16(a1[ds], b, acc);
    }
    const float bias = b1[nt * 16 + l16];
#pragma unroll
    for (int i = 0; i < 8; ++i) {
      float v = acc[i] + bias; v = v > 0.f ? v : 0.f;
      hw[(i + hh * 8) * HS + nt * 16 + l16] = (__bf16)v;   // C layout -> LDS
    }
  }
  __syncthreads();

  // ---- layer 2: h2 = relu(h1 @ w2 + b2), K=128 ----
  FragU a2[4];
#pragma unroll
  for (int f = 0; f < 4; ++f) {
    const int d0 = f * 32 + hh * 8;
    const __bf16* p = hw + l16 * HS + d0;
    a2[f].q[0] = *(const uint4*)p;
    a2[f].q[1] = *(const uint4*)(p + 16);
  }
  __syncthreads();
#pragma unroll
  for (int nt = 0; nt < 8; ++nt) {
    v8f acc = {};
#pragma unroll
    for (int ds = 0; ds < 4; ++ds) {
      FragU b;
      const __bf16* p = w2t + (size_t)(nt * 16 + l16) * HID + ds * 32 + hh * 8;
      b.q[0] = *(const uint4*)p; b.q[1] = *(const uint4*)(p + 16);
      acc = wmma_bf16(a2[ds], b, acc);
    }
    const float bias = b2[nt * 16 + l16];
#pragma unroll
    for (int i = 0; i < 8; ++i) {
      float v = acc[i] + bias; v = v > 0.f ? v : 0.f;
      hw[(i + hh * 8) * HS + nt * 16 + l16] = (__bf16)v;
    }
  }
  __syncthreads();

  // ---- layer 3: out = h2 @ w3 + b3, K=128, N=7 (padded to 16) ----
  FragU a3[4];
#pragma unroll
  for (int f = 0; f < 4; ++f) {
    const int d0 = f * 32 + hh * 8;
    const __bf16* p = hw + l16 * HS + d0;
    a3[f].q[0] = *(const uint4*)p;
    a3[f].q[1] = *(const uint4*)(p + 16);
  }
  v8f acc = {};
#pragma unroll
  for (int ds = 0; ds < 4; ++ds) {
    FragU b;
    const __bf16* p = w3t + (size_t)l16 * HID + ds * 32 + hh * 8;
    b.q[0] = *(const uint4*)p; b.q[1] = *(const uint4*)(p + 16);
    acc = wmma_bf16(a3[ds], b, acc);
  }
  if (l16 < AOUT) {
    const float bias = b3[l16];
#pragma unroll
    for (int i = 0; i < 8; ++i) {
      const size_t grow = rbase + hh * 8 + i;
      out[grow * AOUT + l16] = acc[i] + bias;
    }
  }
}

// ------------------------------- launcher ----------------------------------

extern "C" void kernel_launch(void* const* d_in, const int* in_sizes, int n_in,
                              void* d_out, int out_size, void* d_ws, size_t ws_size,
                              hipStream_t stream) {
  const float* z   = (const float*)d_in[0];
  const float* cbf = (const float*)d_in[1];
  const float* w1  = (const float*)d_in[2];
  const float* b1  = (const float*)d_in[3];
  const float* w2  = (const float*)d_in[4];
  const float* b2  = (const float*)d_in[5];
  const float* w3  = (const float*)d_in[6];
  const float* b3  = (const float*)d_in[7];
  float* out = (float*)d_out;

  char* ws = (char*)d_ws;
  const size_t CB_B  = (size_t)GQ * KC * DIM * 2;   // 4 MB  bf16 codebooks
  const size_t CN_B  = (size_t)GQ * KC * 4;         // 16 KB 0.5*||c||^2
  const size_t W1_B  = (size_t)HID * DIM * 2;       // 128 KB
  const size_t W2_B  = (size_t)HID * HID * 2;       // 32 KB
  const size_t W3_B  = (size_t)16 * HID * 2;        // 4 KB (padded)
  __bf16* cb_bf = (__bf16*)ws;
  float*  cnorm = (float*)(ws + CB_B);
  __bf16* w1t   = (__bf16*)(ws + CB_B + CN_B);
  __bf16* w2t   = (__bf16*)(ws + CB_B + CN_B + W1_B);
  __bf16* w3t   = (__bf16*)(ws + CB_B + CN_B + W1_B + W2_B);
  __bf16* stb   = (__bf16*)(ws + CB_B + CN_B + W1_B + W2_B + W3_B);  // 64 MB

  k_cvt_cb<<<(GQ * KC * DIM) / 256, 256, 0, stream>>>(cbf, cb_bf);
  k_cnorm<<<(GQ * KC) / 256, 256, 0, stream>>>(cbf, cnorm);
  k_cvt_w<<<(HID * DIM + HID * HID + 16 * HID) / 256, 256, 0, stream>>>(
      w1, w2, w3, w1t, w2t, w3t);
  vq_kernel<<<B_ROWS / 128, 256, 0, stream>>>(z, cb_bf, cnorm, stb);
  mlp_kernel<<<B_ROWS / 128, 256, 0, stream>>>(stb, w1t, b1, w2t, b2, w3t, b3, out);
}
